// PolicyGNN_35897336660646
// MI455X (gfx1250) — matrix operand order repaired and checked by reading, not compiled, not run on hardware
//
#include <hip/hip_runtime.h>

// ---------------- problem constants ----------------
#define N_NODES 10000
#define N_EDGES 320000
#define DN 128
#define DE 128
#define DG 64
#define HE 256
#define HN 256
#define KE 448            // 2*DN + DE + DG  (edge MLP input)
#define KN 320            // DN + DE + DG    (node MLP input)
#define KE_T (KE / 32)    // 14 k-tiles
#define KN_T (KN / 32)    // 10 k-tiles

typedef __attribute__((ext_vector_type(16))) __bf16 v16bf;
typedef __attribute__((ext_vector_type(8)))  float  v8f;

union BfFrag { unsigned u[8]; v16bf v; };

__device__ __forceinline__ unsigned short f2bf(float x) {
  unsigned u = __builtin_bit_cast(unsigned, x);
  u = (u + 0x7FFFu + ((u >> 16) & 1u)) >> 16;
  return (unsigned short)u;
}
__device__ __forceinline__ unsigned pk2bf(float a, float b) {
  return (unsigned)f2bf(a) | ((unsigned)f2bf(b) << 16);
}
__device__ __forceinline__ float bf2f(unsigned short h) {
  unsigned u = ((unsigned)h) << 16;
  return __builtin_bit_cast(float, u);
}
__device__ __forceinline__ float lrelu(float x) { return x > 0.f ? x : 0.01f * x; }

#define WAIT_DS() asm volatile("s_wait_dscnt 0x0" ::: "memory")
#define WAIT_ASYNC(imm) asm volatile("s_wait_asynccnt " #imm ::: "memory")

// async global -> LDS 16B copy (GVS mode: saddr base + 32-bit vgpr offset)
__device__ __forceinline__ void async_b128(unsigned lds_byte_off, const void* base,
                                           unsigned glb_byte_off) {
  asm volatile("global_load_async_to_lds_b128 %0, %1, %2 offset:0"
               :: "v"(lds_byte_off), "v"(glb_byte_off), "s"(base)
               : "memory");
}

template <typename T>
__device__ __forceinline__ unsigned lds_off(T* p) {
  return (unsigned)(reinterpret_cast<uintptr_t>(p));
}

// A fragment (16x32 bf16) from a row-major LDS tile.
// Layout: VGPR0..7 lanes0-15 K=0..15 (pairs), lanes16-31 K=16..31 (matches prepacked B).
__device__ __forceinline__ v16bf load_a_frag(const unsigned short* tile, int stride,
                                             int mrow, int half, int kbase) {
  BfFrag f;
  const unsigned short* rp = tile + mrow * stride + kbase;
#pragma unroll
  for (int i = 0; i < 8; ++i) {
    const int k = half * 16 + 2 * i;
    f.u[i] = *(const unsigned*)(rp + k);
  }
  return f.v;
}

// B fragment (32x16 bf16), prepacked so each lane reads 32 contiguous bytes.
__device__ __forceinline__ v16bf load_b_frag(const unsigned* wp, int lane) {
  BfFrag f;
  const uint4* p = (const uint4*)(wp + lane * 8);
  *(uint4*)&f.u[0] = p[0];
  *(uint4*)&f.u[4] = p[1];
  return f.v;
}

__device__ __forceinline__ v8f wmma_bf16(v16bf a, v16bf b, v8f c) {
  return __builtin_amdgcn_wmma_f32_16x16x32_bf16(false, a, false, b, (short)0, c,
                                                 false, false);
}

// ---------------- weight prepack: f32 [K x N] row-major -> per-fragment lane-packed bf16
// B layout used here (same k order as load_a_frag): VGPR i: lanes0-15 -> K = 2i,2i+1 (N = lane); lanes16-31 -> K = 16+2i,17+2i.
__global__ void prepack_w(const float* __restrict__ W, unsigned* __restrict__ out,
                          int K, int Nn) {
  const int idx = blockIdx.x * blockDim.x + threadIdx.x;
  const int NT = Nn / 16;
  const int total = (K / 32) * NT * 256;
  if (idx >= total) return;
  const int i = idx & 7;
  const int lane = (idx >> 3) & 31;
  const int frag = idx >> 8;
  const int nt = frag % NT;
  const int kt = frag / NT;
  const int n = nt * 16 + (lane & 15);
  const int k = kt * 32 + ((lane >> 4) ? 16 : 0) + 2 * i;
  out[idx] = pk2bf(W[k * Nn + n], W[(k + 1) * Nn + n]);
}

// ---------------- embeddings ----------------
__global__ void embed_nodes(const float* __restrict__ x, const float* __restrict__ W,
                            const float* __restrict__ b, unsigned short* __restrict__ xn) {
  const int idx = blockIdx.x * blockDim.x + threadIdx.x;
  if (idx >= N_NODES * DN) return;
  const int n = idx >> 7, f = idx & 127;
  float s = b[f];
  const float* xp = x + n * 8;
#pragma unroll
  for (int k = 0; k < 8; ++k) s += xp[k] * W[k * DN + f];
  xn[idx] = f2bf(lrelu(s));
}

__global__ void embed_edges(const float* __restrict__ ea, const float* __restrict__ W,
                            const float* __restrict__ b, unsigned short* __restrict__ ebuf) {
  const long long idx = (long long)blockIdx.x * blockDim.x + threadIdx.x;
  if (idx >= (long long)N_EDGES * DE) return;
  const int e = (int)(idx >> 7), f = (int)(idx & 127);
  float s = b[f];
  const float* ep = ea + (size_t)e * 4;
#pragma unroll
  for (int k = 0; k < 4; ++k) s += ep[k] * W[k * DE + f];
  ebuf[idx] = f2bf(lrelu(s));
}

__global__ void embed_global(const float* __restrict__ u, const float* __restrict__ W,
                             const float* __restrict__ b, float* gbuf,
                             unsigned short* gbf) {
  const int t = threadIdx.x;  // 64 threads
  float s = b[t];
#pragma unroll
  for (int k = 0; k < 4; ++k) s += u[k] * W[k * DG + t];
  s = lrelu(s);
  gbuf[t] = s;
  gbf[t] = f2bf(s);
}

// ---------------- misc ----------------
__global__ void fill_zero(float* p, int n) {
  const int i = blockIdx.x * blockDim.x + threadIdx.x;
  if (i < n) p[i] = 0.f;
}

__global__ void deg_count(const int* __restrict__ col, float* deg) {
  const int i = blockIdx.x * blockDim.x + threadIdx.x;
  if (i < N_EDGES) atomicAdd(&deg[col[i]], 1.0f);
}

// ============ Edge MLP: 448->256 (lrelu) ->128, fused scatter-add to agg ============
// 4 waves/block, one 16-edge tile per wave. B fragments for each N-tile are staged
// once per block into LDS via async global->LDS copies (double buffered, ASYNCcnt)
// and shared by all 4 waves: 4x cut in L2 weight traffic.
#define EW 4
#define B1_BYTES (KE_T * 1024)   // 14336 B: 14 fragments per N-tile (layer 1)
#define B1_CHUNKS (B1_BYTES / 16)
#define B2_BYTES (8 * 1024)      // 8192 B: 8 fragments per N-tile (layer 2)
#define B2_CHUNKS (B2_BYTES / 16)

__global__ __launch_bounds__(EW * 32) void edge_mlp_kernel(
    const int* __restrict__ row, const int* __restrict__ col,
    const unsigned short* __restrict__ xn, unsigned short* ebuf,
    const unsigned short* __restrict__ gbf,
    const unsigned* __restrict__ w1p, const float* __restrict__ b1,
    const unsigned* __restrict__ w2p, const float* __restrict__ b2,
    float* agg) {
  __shared__ __align__(16) unsigned short sA[EW][16 * KE];
  __shared__ __align__(16) unsigned short sH[EW][16 * HE];
  __shared__ __align__(16) unsigned sB[2][B1_BYTES / 4];
  const int lane = threadIdx.x & 31;
  const int wid = threadIdx.x >> 5;
  const int tile = blockIdx.x * EW + wid;
  const int ebase = tile * 16;
  unsigned short* A = sA[wid];
  unsigned short* H = sH[wid];

  // --- kick async staging of layer-1 B fragments for nt=0 (all 128 threads) ---
  const unsigned sB0 = lds_off(&sB[0][0]);
  const unsigned sB1 = lds_off(&sB[1][0]);
  {
#pragma unroll
    for (int j = 0; j < B1_CHUNKS / (EW * 32); ++j) {
      const int chunk = threadIdx.x + EW * 32 * j;          // 0..895
      const int kt = chunk >> 6, within = chunk & 63;
      async_b128(sB0 + chunk * 16, w1p,
                 (unsigned)(((kt * (HE / 16) + 0) << 10) + (within << 4)));
    }
  }

  // --- gather concat[xn[row], xn[col], e, g] as bf16 into wave-private LDS tile ---
  for (int m = 0; m < 16; ++m) {
    const int eidx = ebase + m;
    const int r = row[eidx];
    const int c = col[eidx];
    ((uint2*)(A + m * KE))[lane]       = ((const uint2*)(xn + (size_t)r * DN))[lane];
    ((uint2*)(A + m * KE + 128))[lane] = ((const uint2*)(xn + (size_t)c * DN))[lane];
    ((uint2*)(A + m * KE + 256))[lane] = ((const uint2*)(ebuf + (size_t)eidx * DE))[lane];
    if (lane < 16) ((uint2*)(A + m * KE + 384))[lane] = ((const uint2*)gbf)[lane];
  }
  WAIT_DS();

  const int mrow = lane & 15, half = lane >> 4;

  // layer 1: keep all 14 A fragments resident in VGPRs
  v16bf a1[KE_T];
#pragma unroll
  for (int kt = 0; kt < KE_T; ++kt) a1[kt] = load_a_frag(A, KE, mrow, half, kt * 32);

#pragma unroll 1
  for (int nt = 0; nt < HE / 16; ++nt) {
    if (nt + 1 < HE / 16) {  // prefetch next N-tile's B into the other buffer
      const unsigned dst = ((nt + 1) & 1) ? sB1 : sB0;
#pragma unroll
      for (int j = 0; j < B1_CHUNKS / (EW * 32); ++j) {
        const int chunk = threadIdx.x + EW * 32 * j;
        const int kt = chunk >> 6, within = chunk & 63;
        async_b128(dst + chunk * 16, w1p,
                   (unsigned)(((kt * (HE / 16) + nt + 1) << 10) + (within << 4)));
      }
      WAIT_ASYNC(0x7);  // drain this wave's 7 copies for the current buffer
    } else {
      WAIT_ASYNC(0x0);
    }
    __syncthreads();  // current buffer fully staged by all waves
    const unsigned* sBf = &sB[nt & 1][0];
    v8f c = {0.f, 0.f, 0.f, 0.f, 0.f, 0.f, 0.f, 0.f};
#pragma unroll
    for (int kt = 0; kt < KE_T; ++kt) {
      v16bf b = load_b_frag(sBf + kt * 256, lane);
      c = wmma_bf16(a1[kt], b, c);
    }
    const float bias = b1[nt * 16 + mrow];
#pragma unroll
    for (int r = 0; r < 8; ++r)
      H[(r + 8 * half) * HE + nt * 16 + mrow] = f2bf(lrelu(c[r] + bias));
    __syncthreads();  // everyone done with this buffer before it is refilled
  }
  WAIT_DS();

  // layer 2
  v16bf a2[8];
#pragma unroll
  for (int kt = 0; kt < 8; ++kt) a2[kt] = load_a_frag(H, HE, mrow, half, kt * 32);

  int myCol[8];
#pragma unroll
  for (int r = 0; r < 8; ++r) myCol[r] = col[ebase + r + 8 * half];

  {  // stage layer-2 B for nt=0
#pragma unroll
    for (int j = 0; j < B2_CHUNKS / (EW * 32); ++j) {
      const int chunk = threadIdx.x + EW * 32 * j;          // 0..511
      const int kt = chunk >> 6, within = chunk & 63;
      async_b128(sB0 + chunk * 16, w2p,
                 (unsigned)(((kt * (DE / 16) + 0) << 10) + (within << 4)));
    }
  }

#pragma unroll 1
  for (int nt = 0; nt < DE / 16; ++nt) {
    if (nt + 1 < DE / 16) {
      const unsigned dst = ((nt + 1) & 1) ? sB1 : sB0;
#pragma unroll
      for (int j = 0; j < B2_CHUNKS / (EW * 32); ++j) {
        const int chunk = threadIdx.x + EW * 32 * j;
        const int kt = chunk >> 6, within = chunk & 63;
        async_b128(dst + chunk * 16, w2p,
                   (unsigned)(((kt * (DE / 16) + nt + 1) << 10) + (within << 4)));
      }
      WAIT_ASYNC(0x4);
    } else {
      WAIT_ASYNC(0x0);
    }
    __syncthreads();
    const unsigned* sBf = &sB[nt & 1][0];
    v8f c = {0.f, 0.f, 0.f, 0.f, 0.f, 0.f, 0.f, 0.f};
#pragma unroll
    for (int kt = 0; kt < 8; ++kt) {
      v16bf b = load_b_frag(sBf + kt * 256, lane);
      c = wmma_bf16(a2[kt], b, c);
    }
    const int f = nt * 16 + mrow;
    const float bias = b2[f];
#pragma unroll
    for (int r = 0; r < 8; ++r) {
      const float v = c[r] + bias;
      H[(r + 8 * half) * DE + f] = f2bf(v);  // stage out-tile for coalesced writeback
      atomicAdd(&agg[(size_t)myCol[r] * DE + f], v);  // fused scatter-sum
    }
    __syncthreads();
  }
  WAIT_DS();

  for (int m = 0; m < 16; ++m)
    ((uint2*)(ebuf + (size_t)(ebase + m) * DE))[lane] = ((const uint2*)(H + m * DE))[lane];
}

// ---------------- Node MLP: 320->256 (lrelu) ->128, fused mean-reduction -----------
#define NW 2  // waves per block, wave-private tiles, no cross-wave barriers

__global__ __launch_bounds__(NW * 32) void node_mlp_kernel(
    const float* __restrict__ agg, const float* __restrict__ deg,
    unsigned short* xn, const unsigned short* __restrict__ gbf,
    const unsigned* __restrict__ w1p, const float* __restrict__ b1,
    const unsigned* __restrict__ w2p, const float* __restrict__ b2,
    float* xsum) {
  __shared__ __align__(16) unsigned short sA[NW][16 * KN];
  __shared__ __align__(16) unsigned short sH[NW][16 * HN];
  const int lane = threadIdx.x & 31;
  const int wid = threadIdx.x >> 5;
  const int tile = blockIdx.x * NW + wid;
  if (tile >= N_NODES / 16) return;  // wave-uniform early out
  const int nbase = tile * 16;
  unsigned short* A = sA[wid];
  unsigned short* H = sH[wid];

  for (int m = 0; m < 16; ++m) {
    const int node = nbase + m;
    ((uint2*)(A + m * KN))[lane] = ((const uint2*)(xn + (size_t)node * DN))[lane];
    const float dinv = 1.0f / fmaxf(deg[node], 1.0f);
    const float2* ap = (const float2*)(agg + (size_t)node * DE);
    unsigned* dst = (unsigned*)(A + m * KN + 128);
#pragma unroll
    for (int j = 0; j < 2; ++j) {
      const float2 t = ap[lane * 2 + j];
      dst[lane * 2 + j] = pk2bf(t.x * dinv, t.y * dinv);
    }
    if (lane < 16) ((uint2*)(A + m * KN + 256))[lane] = ((const uint2*)gbf)[lane];
  }
  WAIT_DS();

  const int mrow = lane & 15, half = lane >> 4;

  v16bf a1[KN_T];
#pragma unroll
  for (int kt = 0; kt < KN_T; ++kt) a1[kt] = load_a_frag(A, KN, mrow, half, kt * 32);

#pragma unroll 1
  for (int nt = 0; nt < HN / 16; ++nt) {
    v8f c = {0.f, 0.f, 0.f, 0.f, 0.f, 0.f, 0.f, 0.f};
#pragma unroll
    for (int kt = 0; kt < KN_T; ++kt) {
      v16bf b = load_b_frag(w1p + (size_t)(kt * (HN / 16) + nt) * 256, lane);
      c = wmma_bf16(a1[kt], b, c);
    }
    const float bias = b1[nt * 16 + mrow];
#pragma unroll
    for (int r = 0; r < 8; ++r)
      H[(r + 8 * half) * HN + nt * 16 + mrow] = f2bf(lrelu(c[r] + bias));
  }
  WAIT_DS();

  v16bf a2[8];
#pragma unroll
  for (int kt = 0; kt < 8; ++kt) a2[kt] = load_a_frag(H, HN, mrow, half, kt * 32);

#pragma unroll 1
  for (int nt = 0; nt < DN / 16; ++nt) {
    v8f c = {0.f, 0.f, 0.f, 0.f, 0.f, 0.f, 0.f, 0.f};
#pragma unroll
    for (int kt = 0; kt < 8; ++kt) {
      v16bf b = load_b_frag(w2p + (size_t)(kt * (DN / 16) + nt) * 256, lane);
      c = wmma_bf16(a2[kt], b, c);
    }
    const int f = nt * 16 + mrow;
    const float bias = b2[f];
#pragma unroll
    for (int r = 0; r < 8; ++r) {
      const float v = c[r] + bias;
      H[(r + 8 * half) * DN + f] = f2bf(v);
      atomicAdd(&xsum[f], v);  // fused node-mean numerator
    }
  }
  WAIT_DS();

  for (int m = 0; m < 16; ++m)
    ((uint2*)(xn + (size_t)(nbase + m) * DN))[lane] = ((const uint2*)(H + m * DN))[lane];
}

// ---------------- Global MLP: [g(64) | mean_nodes(128)] -> 256 -> 64 --------------
__global__ void global_mlp(const float* __restrict__ xsum, float* gbuf,
                           unsigned short* gbf, const float* __restrict__ W1,
                           const float* __restrict__ bb1, const float* __restrict__ W2,
                           const float* __restrict__ bb2) {
  __shared__ float a[DG + DN];
  __shared__ float h[256];
  const int t = threadIdx.x;  // 256 threads
  if (t < DG) a[t] = gbuf[t];
  else if (t < DG + DN) a[t] = xsum[t - DG] * (1.0f / (float)N_NODES);
  __syncthreads();
  float s = bb1[t];
  for (int k = 0; k < DG + DN; ++k) s += a[k] * W1[k * 256 + t];
  h[t] = lrelu(s);
  __syncthreads();
  if (t < DG) {
    float o = bb2[t];
    for (int j = 0; j < 256; ++j) o += h[j] * W2[j * DG + t];
    gbuf[t] = o;
    gbf[t] = f2bf(o);
  }
}

// ---------------- decoder: out[e] = e_row . W_dec + b, one wave per edge ----------
__global__ void decoder(const unsigned short* __restrict__ ebuf,
                        const float* __restrict__ wdec, const float* __restrict__ bdec,
                        float* __restrict__ out) {
  const int gid = blockIdx.x * blockDim.x + threadIdx.x;
  const int e = gid >> 5;
  const int lane = gid & 31;
  if (e >= N_EDGES) return;
  const unsigned short* ep = ebuf + (size_t)e * DE;
  float s = 0.f;
#pragma unroll
  for (int j = 0; j < 4; ++j) {
    const int k = lane * 4 + j;
    s += bf2f(ep[k]) * wdec[k];
  }
#pragma unroll
  for (int off = 16; off; off >>= 1) s += __shfl_xor(s, off, 32);
  if (lane == 0) out[e] = s + bdec[0];
}

// ---------------- host ----------------
extern "C" void kernel_launch(void* const* d_in, const int* in_sizes, int n_in,
                              void* d_out, int out_size, void* d_ws, size_t ws_size,
                              hipStream_t stream) {
  (void)in_sizes; (void)n_in; (void)out_size; (void)ws_size;
  const float* x    = (const float*)d_in[0];
  const float* eatt = (const float*)d_in[1];
  const float* u    = (const float*)d_in[2];
  const int* eidx   = (const int*)d_in[3];
  const int* row = eidx;
  const int* col = eidx + N_EDGES;
  const float* W_ne = (const float*)d_in[5],  *b_ne = (const float*)d_in[6];
  const float* W_ee = (const float*)d_in[7],  *b_ee = (const float*)d_in[8];
  const float* W_ge = (const float*)d_in[9],  *b_ge = (const float*)d_in[10];
  const float* W_e1 = (const float*)d_in[11], *b_e1 = (const float*)d_in[12];
  const float* W_e2 = (const float*)d_in[13], *b_e2 = (const float*)d_in[14];
  const float* W_n1 = (const float*)d_in[15], *b_n1 = (const float*)d_in[16];
  const float* W_n2 = (const float*)d_in[17], *b_n2 = (const float*)d_in[18];
  const float* W_g1 = (const float*)d_in[19], *b_g1 = (const float*)d_in[20];
  const float* W_g2 = (const float*)d_in[21], *b_g2 = (const float*)d_in[22];
  const float* W_dec = (const float*)d_in[23], *b_dec = (const float*)d_in[24];

  char* p = (char*)d_ws;
  auto carve = [&](size_t bytes) {
    void* r = (void*)p;
    p += (bytes + 255) & ~(size_t)255;
    return r;
  };
  unsigned short* xn   = (unsigned short*)carve((size_t)N_NODES * DN * 2);
  unsigned short* ebuf = (unsigned short*)carve((size_t)N_EDGES * DE * 2);
  unsigned short* gbf  = (unsigned short*)carve(DG * 2);
  float* gbuf = (float*)carve(DG * 4);
  float* agg  = (float*)carve((size_t)N_NODES * DE * 4);
  float* deg  = (float*)carve((size_t)N_NODES * 4);
  float* xsum = (float*)carve(DN * 4);
  unsigned* w1e = (unsigned*)carve((size_t)KE_T * (HE / 16) * 1024);
  unsigned* w2e = (unsigned*)carve((size_t)8 * (DE / 16) * 1024);
  unsigned* w1n = (unsigned*)carve((size_t)KN_T * (HN / 16) * 1024);
  unsigned* w2n = (unsigned*)carve((size_t)8 * (DN / 16) * 1024);

  auto prepack = [&](const float* W, unsigned* o, int K, int Nn) {
    const int total = (K / 32) * (Nn / 16) * 256;
    prepack_w<<<(total + 255) / 256, 256, 0, stream>>>(W, o, K, Nn);
  };
  prepack(W_e1, w1e, KE, HE);
  prepack(W_e2, w2e, HE, DE);
  prepack(W_n1, w1n, KN, HN);
  prepack(W_n2, w2n, HN, DN);

  embed_nodes<<<(N_NODES * DN + 255) / 256, 256, 0, stream>>>(x, W_ne, b_ne, xn);
  embed_edges<<<(int)(((long long)N_EDGES * DE + 255) / 256), 256, 0, stream>>>(eatt, W_ee, b_ee, ebuf);
  embed_global<<<1, 64, 0, stream>>>(u, W_ge, b_ge, gbuf, gbf);

  fill_zero<<<(N_NODES + 255) / 256, 256, 0, stream>>>(deg, N_NODES);
  deg_count<<<(N_EDGES + 255) / 256, 256, 0, stream>>>(col, deg);

  for (int pass = 0; pass < 2; ++pass) {
    fill_zero<<<(N_NODES * DE + 255) / 256, 256, 0, stream>>>(agg, N_NODES * DE);
    fill_zero<<<1, 128, 0, stream>>>(xsum, DN);
    edge_mlp_kernel<<<N_EDGES / 16 / EW, EW * 32, 0, stream>>>(
        row, col, xn, ebuf, gbf, w1e, b_e1, w2e, b_e2, agg);
    node_mlp_kernel<<<(N_NODES / 16 + NW - 1) / NW, NW * 32, 0, stream>>>(
        agg, deg, xn, gbf, w1n, b_n1, w2n, b_n2, xsum);
    global_mlp<<<1, 256, 0, stream>>>(xsum, gbuf, gbf, W_g1, b_g1, W_g2, b_g2);
  }

  decoder<<<(N_EDGES * 32 + 255) / 256, 256, 0, stream>>>(ebuf, W_dec, b_dec,
                                                          (float*)d_out);
}
